// ClassLoss_11828339933550
// MI455X (gfx1250) — compile-verified, matrix operand order checked by reading, not run on hardware
//
#include <hip/hip_runtime.h>
#include <hip/hip_bf16.h>
#include <math.h>

typedef __attribute__((ext_vector_type(16))) _Float16 v16h;
typedef __attribute__((ext_vector_type(8)))  float    v8f;

#define T_TGT  100
#define NANCH  3
#define NCLS   80
#define NBATCH 16

__global__ void zero_out_kernel(float* out) { out[0] = 0.0f; }

// One wave32 block per (scale, batch). Sparse-gather CE loss:
// decode+dedup targets, then logsumexp over gathered 80-wide rows,
// 16 rows per WMMA group (exp-sum reduced via v_wmma_f32_16x16x32_f16
// against an all-ones B matrix -> f32 row sums).
__global__ __launch_bounds__(32) void yolo_class_loss_kernel(
    const float* __restrict__ out0,
    const float* __restrict__ out1,
    const float* __restrict__ out2,
    const float* __restrict__ targets,
    float* __restrict__ loss_out)
{
    const int lane = threadIdx.x;            // 0..31
    const int s    = blockIdx.x / NBATCH;    // scale 0..2
    const int b    = blockIdx.x % NBATCH;    // batch

    int H;
    const float* base;
    if (s == 0)      { H = 128; base = out0; }
    else if (s == 1) { H = 64;  base = out1; }
    else             { H = 32;  base = out2; }
    const int W = H;

    __shared__ int sR[T_TGT], sC[T_TGT], sK[T_TGT], sInr[T_TGT], sKeep[T_TGT];
    __shared__ int cellR[T_TGT], cellC[T_TGT], cellK[T_TGT];
    __shared__ int sM;
    __shared__ float sums[16 * 16];

    // ---- decode targets (parallel over lanes) ----
    for (int t = lane; t < T_TGT; t += 32) {
        const float* tg = targets + ((size_t)b * T_TGT + t) * 5;
        float f0 = tg[0], f1 = tg[1], f2 = tg[2], f3 = tg[3], f4 = tg[4];
        bool valid = !(f0 == 0.f && f1 == 0.f && f2 == 0.f && f3 == 0.f && f4 == 0.f);
        int r = valid ? (int)(f2 * (float)H) : H;   // invalid -> OOB row -> dropped
        int c = (int)(f1 * (float)W);
        int k = (int)f0;
        sR[t] = r; sC[t] = c; sK[t] = k;
        sInr[t] = (valid && r >= 0 && r < H && c >= 0 && c < W) ? 1 : 0;
    }
    __syncthreads();

    // ---- dedup: last scatter to a cell wins ----
    for (int t = lane; t < T_TGT; t += 32) {
        int keep = sInr[t];
        if (keep) {
            for (int u = t + 1; u < T_TGT; ++u)
                if (sInr[u] && sR[u] == sR[t] && sC[u] == sC[t]) { keep = 0; break; }
        }
        sKeep[t] = keep;
    }
    __syncthreads();
    if (lane == 0) {
        int M = 0;
        for (int t = 0; t < T_TGT; ++t)
            if (sKeep[t]) { cellR[M] = sR[t]; cellC[M] = sC[t]; cellK[M] = sK[t]; ++M; }
        sM = M;
    }
    __syncthreads();

    const int M       = sM;
    const int nrows   = NANCH * M;
    const int ngroups = (nrows + 15) >> 4;    // wave-uniform
    const int h   = lane >> 4;                // which half of the row this lane owns
    const int m16 = lane & 15;                // row-in-group

    v16h bones;
#pragma unroll
    for (int e = 0; e < 16; ++e) bones[e] = (_Float16)1.0f;

    float acc_nll = 0.0f;

    for (int g = 0; g < ngroups; ++g) {
        int  row    = g * 16 + m16;
        bool rvalid = row < nrows;
        int  row_s  = rvalid ? row : 0;       // dummy rows -> cell 0 (rows don't mix in WMMA)
        int  j = row_s / NANCH;
        int  a = row_s - NANCH * j;
        int  r = cellR[j], c = cellC[j], kcls = cellK[j];
        size_t i = (size_t)(r * W + c) * NANCH + a;
        const float* rowp = base + ((size_t)b * NANCH * H * W + i) * 85 + 5;

        // Each lane owns the 40 columns matching its WMMA A-layout slots:
        // chunk gc covers K = 32*gc + kl, kl = (e<8 ? e : e+8) + 8*h.
        float vals[3][16];
        float lmax = -INFINITY;
#pragma unroll
        for (int gc = 0; gc < 3; ++gc) {
#pragma unroll
            for (int e = 0; e < 16; ++e) {
                int kl = (e < 8) ? (e + 8 * h) : (e + 8 + 8 * h);
                int kg = 32 * gc + kl;
                float v = (kg < NCLS) ? rowp[kg] : -INFINITY;
                vals[gc][e] = v;
                lmax = fmaxf(lmax, v);
            }
        }
        float rmax = fmaxf(lmax, __shfl_xor(lmax, 16, 32));  // full-row max

        // exp(x - max) -> f16 A tiles, row-sum via WMMA with ones-B (f32 accum)
        v8f accv = {};
#pragma unroll
        for (int gc = 0; gc < 3; ++gc) {
            v16h am;
#pragma unroll
            for (int e = 0; e < 16; ++e) {
                float v  = vals[gc][e];
                float ev = (v == -INFINITY) ? 0.0f : __expf(v - rmax);
                am[e] = (_Float16)ev;
            }
            accv = __builtin_amdgcn_wmma_f32_16x16x32_f16(
                false, am, false, bones, (short)0, accv, false, false);
        }

        // D[m][n]: lane L vgpr v -> m = v + 8*(L/16), n = L%16
#pragma unroll
        for (int v = 0; v < 8; ++v)
            sums[(v + 8 * h) * 16 + m16] = accv[v];
        __syncthreads();

        if (lane < 16 && rvalid) {
            float S  = sums[lane * 16 + 0];          // row sum of exp
            float xc = rowp[kcls];
            acc_nll += (rmax + __logf(S)) - xc;      // logsumexp - x[cls]
        }
        __syncthreads();                             // protect sums[] reuse
    }

    // wave-level sum of per-lane nll contributions
#pragma unroll
    for (int off = 16; off >= 1; off >>= 1)
        acc_nll += __shfl_xor(acc_nll, off, 32);

    if (lane == 0) {
        int denom = (NANCH * M > 1) ? (NANCH * M) : 1;
        atomicAdd(loss_out, (acc_nll / (float)denom) * (1.0f / (float)NBATCH));
    }
}

extern "C" void kernel_launch(void* const* d_in, const int* in_sizes, int n_in,
                              void* d_out, int out_size, void* d_ws, size_t ws_size,
                              hipStream_t stream) {
    const float* out0    = (const float*)d_in[0];
    const float* out1    = (const float*)d_in[1];
    const float* out2    = (const float*)d_in[2];
    const float* targets = (const float*)d_in[3];
    float* loss = (float*)d_out;

    zero_out_kernel<<<1, 1, 0, stream>>>(loss);
    yolo_class_loss_kernel<<<3 * NBATCH, 32, 0, stream>>>(out0, out1, out2, targets, loss);
}
